// GemmaExpert_58660663328847
// MI455X (gfx1250) — compile-verified, matrix operand order checked by reading, not run on hardware
//
#include <hip/hip_runtime.h>
#include <stdint.h>

// ---------------------------------------------------------------------------
// Types for CDNA5 WMMA (wave32): v_wmma_f32_16x16x32_bf16
// ---------------------------------------------------------------------------
typedef __bf16 bf16;
typedef __bf16 v8bf  __attribute__((ext_vector_type(8)));
typedef __bf16 v16bf __attribute__((ext_vector_type(16)));
typedef float  v8f   __attribute__((ext_vector_type(8)));

#define USE_ASYNC_LDS 1

namespace cfg {
constexpr int B  = 2, S = 2048, H = 2048;
constexpr int NH = 4, NKV = 2, HD = 256, IP = 4096;
constexpr int T  = B * S;        // 4096 tokens
constexpr int QD = NH * HD;      // 1024
constexpr int KD = NKV * HD;     // 512
}

__device__ __forceinline__ v8f wmma_bf16(v16bf a, v16bf b, v8f c) {
  // (neg_a, A, neg_b, B, c_mod, C, reuse_a, reuse_b)
  return __builtin_amdgcn_wmma_f32_16x16x32_bf16(false, a, false, b, (short)0, c,
                                                 false, false);
}

#if USE_ASYNC_LDS
__device__ __forceinline__ void async_b128(uint32_t lds_off, const void* gaddr) {
  asm volatile("global_load_async_to_lds_b128 %0, %1, off"
               :: "v"(lds_off), "v"((unsigned long long)(uintptr_t)gaddr)
               : "memory");
}
__device__ __forceinline__ void async_wait_all() {
  asm volatile("s_wait_asynccnt 0" ::: "memory");
}
#else
__device__ __forceinline__ void async_wait_all() {}
#endif

// Stage one 16-byte chunk global -> LDS.
__device__ __forceinline__ void stage16(bf16* lds_dst, const bf16* gsrc) {
#if USE_ASYNC_LDS
  async_b128((uint32_t)(uintptr_t)lds_dst, gsrc);
#else
  *(v8bf*)lds_dst = *(const v8bf*)gsrc;
#endif
}

// Load one 16x32 bf16 fragment (A or B operand) from an LDS tile with leading
// dim `ld` elements, following the documented 16-bit operand lane layout:
//   lanes 0-15 : row = lane,    K = {0..7, 16..23}
//   lanes 16-31: row = lane-16, K = {8..15, 24..31}
__device__ __forceinline__ v16bf load_frag(const bf16* lds, int row0, int ld,
                                           int lane) {
  const int r  = lane & 15;
  const int ko = (lane >> 4) * 8;
  const bf16* base = lds + (row0 + r) * ld + ko;
  v8bf lo = *(const v8bf*)(base);        // ds_load_b128
  v8bf hi = *(const v8bf*)(base + 16);   // ds_load_b128
  v16bf f;
#pragma unroll
  for (int i = 0; i < 8; ++i) { f[i] = lo[i]; f[i + 8] = hi[i]; }
  return f;
}

// ---------------------------------------------------------------------------
// C[M,N] (+)= A[M,K] * B[N,K]^T    (bf16 in, f32 out)
// block = 256 threads = 8 waves; block tile 128x256; wave tile 64x64; BK=32.
// Double-buffered LDS with async global->LDS prefetch of the next K tile.
// M % 128 == 0, N % 256 == 0, K % 32 == 0 at every call site.
// ---------------------------------------------------------------------------
template <bool ACC>
__global__ __launch_bounds__(256) void gemm_nt(
    const bf16* __restrict__ A, int lda,
    const bf16* __restrict__ B, int ldb,
    float* __restrict__ C, int ldc, int K) {
  constexpr int BM = 128, BN = 256, BK = 32;
  __shared__ bf16 As[2][BM * BK];
  __shared__ bf16 Bs[2][BN * BK];

  const int tid  = threadIdx.x;
  const int wave = tid >> 5;
  const int lane = tid & 31;
  const int wm   = (wave & 1) * 64;   // wave grid: 2 (M) x 4 (N)
  const int wn   = (wave >> 1) * 64;
  const long bm  = (long)blockIdx.y * BM;
  const long bn  = (long)blockIdx.x * BN;

  // staging: A = 512 chunks (2/thread), B = 1024 chunks (4/thread)
  const int crow  = tid >> 2;         // 0..63
  const int cpart = (tid & 3) * 8;    // element offset within the 32-col row

  auto stage_tile = [&](int buf, int k0) {
#pragma unroll
    for (int h = 0; h < 2; ++h) {
      const int row = crow + h * 64;
      stage16(&As[buf][row * BK + cpart],
              A + (bm + row) * (long)lda + k0 + cpart);
    }
#pragma unroll
    for (int h = 0; h < 4; ++h) {
      const int row = crow + h * 64;
      stage16(&Bs[buf][row * BK + cpart],
              B + (bn + row) * (long)ldb + k0 + cpart);
    }
  };

  v8f acc[4][4] = {};

  stage_tile(0, 0);
  async_wait_all();
  __syncthreads();

  int cur = 0;
  for (int k0 = 0; k0 < K; k0 += BK) {
    if (k0 + BK < K) stage_tile(cur ^ 1, k0 + BK);  // async prefetch next tile

    v16bf af[4], bfr[4];
#pragma unroll
    for (int i = 0; i < 4; ++i) af[i] = load_frag(As[cur], wm + 16 * i, BK, lane);
#pragma unroll
    for (int j = 0; j < 4; ++j) bfr[j] = load_frag(Bs[cur], wn + 16 * j, BK, lane);
#pragma unroll
    for (int i = 0; i < 4; ++i)
#pragma unroll
      for (int j = 0; j < 4; ++j)
        acc[i][j] = wmma_bf16(af[i], bfr[j], acc[i][j]);

    async_wait_all();   // prefetch landed
    __syncthreads();    // everyone done reading `cur`
    cur ^= 1;
  }

  // f32 16x16 C/D layout: lane&15 -> col, VGPR r -> row r + 8*(lane>=16)
  const int col = lane & 15;
  const int hi  = lane >> 4;
#pragma unroll
  for (int i = 0; i < 4; ++i)
#pragma unroll
    for (int j = 0; j < 4; ++j)
#pragma unroll
      for (int r = 0; r < 8; ++r) {
        const long row = bm + wm + 16 * i + r + 8 * hi;
        const long cc  = bn + wn + 16 * j + col;
        if (ACC) C[row * ldc + cc] += acc[i][j][r];
        else     C[row * ldc + cc]  = acc[i][j][r];
      }
}

// ---------------------------------------------------------------------------
// Fused gate/up GEMM + SwiGLU epilogue: H = silu(A G^T) * (A U^T)  (bf16 out)
// block tile 128x128; 8 waves (4 M x 2 N), wave tile 32x64; dual accumulators
// share the A tile. Double-buffered async staging.
// ---------------------------------------------------------------------------
__global__ __launch_bounds__(256) void gemm_swiglu(
    const bf16* __restrict__ A, int lda,
    const bf16* __restrict__ G, const bf16* __restrict__ U, int ldb,
    bf16* __restrict__ Hm, int ldh, int K) {
  constexpr int BM = 128, BN = 128, BK = 32;
  __shared__ bf16 As[2][BM * BK];
  __shared__ bf16 Gs[2][BN * BK];
  __shared__ bf16 Us[2][BN * BK];

  const int tid  = threadIdx.x;
  const int wave = tid >> 5;
  const int lane = tid & 31;
  const int wm   = (wave >> 1) * 32;  // 4 waves along M
  const int wn   = (wave & 1) * 64;   // 2 waves along N
  const long bm  = (long)blockIdx.y * BM;
  const long bn  = (long)blockIdx.x * BN;

  const int crow  = tid >> 2;         // 0..63
  const int cpart = (tid & 3) * 8;

  auto stage_tile = [&](int buf, int k0) {
#pragma unroll
    for (int h = 0; h < 2; ++h) {
      const int row = crow + h * 64;
      stage16(&As[buf][row * BK + cpart],
              A + (bm + row) * (long)lda + k0 + cpart);
      stage16(&Gs[buf][row * BK + cpart],
              G + (bn + row) * (long)ldb + k0 + cpart);
      stage16(&Us[buf][row * BK + cpart],
              U + (bn + row) * (long)ldb + k0 + cpart);
    }
  };

  v8f cg[2][4] = {}, cu[2][4] = {};

  stage_tile(0, 0);
  async_wait_all();
  __syncthreads();

  int cur = 0;
  for (int k0 = 0; k0 < K; k0 += BK) {
    if (k0 + BK < K) stage_tile(cur ^ 1, k0 + BK);

    v16bf af[2], gf[4], uf[4];
#pragma unroll
    for (int i = 0; i < 2; ++i) af[i] = load_frag(As[cur], wm + 16 * i, BK, lane);
#pragma unroll
    for (int j = 0; j < 4; ++j) {
      gf[j] = load_frag(Gs[cur], wn + 16 * j, BK, lane);
      uf[j] = load_frag(Us[cur], wn + 16 * j, BK, lane);
    }
#pragma unroll
    for (int i = 0; i < 2; ++i)
#pragma unroll
      for (int j = 0; j < 4; ++j) {
        cg[i][j] = wmma_bf16(af[i], gf[j], cg[i][j]);
        cu[i][j] = wmma_bf16(af[i], uf[j], cu[i][j]);
      }

    async_wait_all();
    __syncthreads();
    cur ^= 1;
  }

  const int col = lane & 15;
  const int hi  = lane >> 4;
#pragma unroll
  for (int i = 0; i < 2; ++i)
#pragma unroll
    for (int j = 0; j < 4; ++j)
#pragma unroll
      for (int r = 0; r < 8; ++r) {
        const long row = bm + wm + 16 * i + r + 8 * hi;
        const long cc  = bn + wn + 16 * j + col;
        const float g = cg[i][j][r];
        const float u = cu[i][j][r];
        // silu(g)*u with hardware reciprocal (bf16 output precision)
        const float sig = __builtin_amdgcn_rcpf(1.0f + __expf(-g));
        Hm[row * (long)ldh + cc] = (bf16)(g * sig * u);
      }
}

// ---------------------------------------------------------------------------
// Elementwise f32 -> bf16
// ---------------------------------------------------------------------------
__global__ __launch_bounds__(256) void cvt_f32_bf16(const float* __restrict__ s,
                                                    bf16* __restrict__ d,
                                                    long n) {
  long i = (long)blockIdx.x * 256 + threadIdx.x;
  const long stride = (long)gridDim.x * 256;
  for (; i < n; i += stride) d[i] = (bf16)s[i];
}

// ---------------------------------------------------------------------------
// RoPE + head packing. One block per token (256 threads = hd lanes).
//   qb: [B,NH,S,HD] bf16  (rotated)
//   kb: [B,NKV,S,HD] bf16 (rotated)
//   vT: [B,NKV,HD,S] bf16 (transposed, so P@V maps onto the NT GEMM)
// ---------------------------------------------------------------------------
__global__ __launch_bounds__(256) void rope_pack(
    const float* __restrict__ qf, const float* __restrict__ kf,
    const float* __restrict__ vf, const float* __restrict__ cosT,
    const float* __restrict__ sinT, bf16* __restrict__ qb,
    bf16* __restrict__ kb, bf16* __restrict__ vT) {
  using namespace cfg;
  const int tok  = blockIdx.x;   // 0..T-1
  const int b    = tok / S;
  const int s    = tok % S;
  const int d    = threadIdx.x;  // 0..HD-1
  const int half = HD / 2;
  const float c  = cosT[(long)s * HD + d];
  const float sn = sinT[(long)s * HD + d];
#pragma unroll
  for (int h = 0; h < NH; ++h) {
    const float* row = qf + (long)tok * QD + h * HD;
    const float x  = row[d];
    const float xr = (d < half) ? -row[d + half] : row[d - half];
    qb[(((long)(b * NH + h)) * S + s) * HD + d] = (bf16)(x * c + xr * sn);
  }
#pragma unroll
  for (int h = 0; h < NKV; ++h) {
    const float* row = kf + (long)tok * KD + h * HD;
    const float x  = row[d];
    const float xr = (d < half) ? -row[d + half] : row[d - half];
    kb[(((long)(b * NKV + h)) * S + s) * HD + d] = (bf16)(x * c + xr * sn);
    const float v = vf[(long)tok * KD + h * HD + d];
    vT[(((long)(b * NKV + h)) * HD + d) * S + s] = (bf16)v;
  }
}

// ---------------------------------------------------------------------------
// Row-wise causal softmax: P[r,c] = softmax(scale * Sc[r, 0..r]); 0 beyond.
// One block per row.
// ---------------------------------------------------------------------------
__global__ __launch_bounds__(256) void softmax_causal(
    const float* __restrict__ Sc, bf16* __restrict__ P, int Sdim, float scale) {
  __shared__ float red[256];
  const int row = blockIdx.x;
  const int tid = threadIdx.x;
  const int n   = row + 1;
  const float* srow = Sc + (long)row * Sdim;

  float m = -1e30f;
  for (int c = tid; c < n; c += 256) m = fmaxf(m, srow[c] * scale);
  red[tid] = m;
  __syncthreads();
  for (int o = 128; o > 0; o >>= 1) {
    if (tid < o) red[tid] = fmaxf(red[tid], red[tid + o]);
    __syncthreads();
  }
  m = red[0];
  __syncthreads();

  float sum = 0.0f;
  for (int c = tid; c < n; c += 256) sum += __expf(srow[c] * scale - m);
  red[tid] = sum;
  __syncthreads();
  for (int o = 128; o > 0; o >>= 1) {
    if (tid < o) red[tid] += red[tid + o];
    __syncthreads();
  }
  const float inv = __builtin_amdgcn_rcpf(red[0]);

  bf16* prow = P + (long)row * Sdim;
  for (int c = tid; c < Sdim; c += 256) {
    const float p = (c < n) ? __expf(srow[c] * scale - m) * inv : 0.0f;
    prow[c] = (bf16)p;
  }
}

// ---------------------------------------------------------------------------
// Host orchestration
// ---------------------------------------------------------------------------
extern "C" void kernel_launch(void* const* d_in, const int* in_sizes, int n_in,
                              void* d_out, int out_size, void* d_ws,
                              size_t ws_size, hipStream_t stream) {
  using namespace cfg;
  (void)in_sizes; (void)n_in; (void)out_size; (void)ws_size;

  const float* x      = (const float*)d_in[0];
  const float* cosT   = (const float*)d_in[1];
  const float* sinT   = (const float*)d_in[2];
  const float* q_w    = (const float*)d_in[3];
  const float* k_w    = (const float*)d_in[4];
  const float* v_w    = (const float*)d_in[5];
  const float* o_w    = (const float*)d_in[6];
  const float* gate_w = (const float*)d_in[7];
  const float* up_w   = (const float*)d_in[8];
  const float* down_w = (const float*)d_in[9];
  float* out = (float*)d_out;

  char* wsp = (char*)d_ws;
  auto alloc = [&](size_t bytes) -> char* {
    char* p = wsp;
    wsp += (bytes + 255) & ~(size_t)255;
    return p;
  };
  bf16* xb   = (bf16*)alloc((size_t)T * H * 2);
  bf16* qwb  = (bf16*)alloc((size_t)QD * H * 2);
  bf16* kwb  = (bf16*)alloc((size_t)KD * H * 2);
  bf16* vwb  = (bf16*)alloc((size_t)KD * H * 2);
  bf16* owb  = (bf16*)alloc((size_t)H * QD * 2);
  bf16* gwb  = (bf16*)alloc((size_t)IP * H * 2);
  bf16* uwb  = (bf16*)alloc((size_t)IP * H * 2);
  bf16* dwb  = (bf16*)alloc((size_t)H * IP * 2);
  float* qf  = (float*)alloc((size_t)T * QD * 4);
  float* kf  = (float*)alloc((size_t)T * KD * 4);
  float* vf  = (float*)alloc((size_t)T * KD * 4);
  bf16* qb   = (bf16*)alloc((size_t)T * QD * 2);
  bf16* kb   = (bf16*)alloc((size_t)T * KD * 2);
  bf16* vT   = (bf16*)alloc((size_t)T * KD * 2);
  float* scf = (float*)alloc((size_t)S * S * 4);
  bf16* pb   = (bf16*)alloc((size_t)S * S * 2);
  float* atf = (float*)alloc((size_t)T * QD * 4);
  bf16* atb  = (bf16*)alloc((size_t)T * QD * 2);
  bf16* hb   = (bf16*)alloc((size_t)T * IP * 2);

  const dim3 blk(256);
  auto cvt = [&](const float* s, bf16* d, long n) {
    cvt_f32_bf16<<<dim3(2048), blk, 0, stream>>>(s, d, n);
  };
  auto gemm = [&](const bf16* A, int lda, const bf16* Bm, int ldb, float* C,
                  int ldc, long M, long N, long K, bool accFlag) {
    dim3 grid((unsigned)(N / 256), (unsigned)(M / 128));
    if (accFlag)
      gemm_nt<true><<<grid, blk, 0, stream>>>(A, lda, Bm, ldb, C, ldc, (int)K);
    else
      gemm_nt<false><<<grid, blk, 0, stream>>>(A, lda, Bm, ldb, C, ldc, (int)K);
  };

  // 1) bf16 conversions
  cvt(x, xb, (long)T * H);
  cvt(q_w, qwb, (long)QD * H);
  cvt(k_w, kwb, (long)KD * H);
  cvt(v_w, vwb, (long)KD * H);
  cvt(o_w, owb, (long)H * QD);
  cvt(gate_w, gwb, (long)IP * H);
  cvt(up_w, uwb, (long)IP * H);
  cvt(down_w, dwb, (long)H * IP);

  // 2) QKV projections
  gemm(xb, H, qwb, H, qf, QD, T, QD, H, false);
  gemm(xb, H, kwb, H, kf, KD, T, KD, H, false);
  gemm(xb, H, vwb, H, vf, KD, T, KD, H, false);

  // 3) RoPE + pack (q/k rotated, v transposed)
  rope_pack<<<dim3(T), blk, 0, stream>>>(qf, kf, vf, cosT, sinT, qb, kb, vT);

  // 4) attention per (batch, head); GQA: head h uses kv head h/2
  const float scale = 1.0f / 16.0f;  // 1/sqrt(256)
  for (int b = 0; b < B; ++b) {
    for (int h = 0; h < NH; ++h) {
      const int kv = h >> 1;
      const bf16* Aq = qb + ((long)(b * NH + h) * S) * HD;
      const bf16* Bk = kb + ((long)(b * NKV + kv) * S) * HD;
      gemm(Aq, HD, Bk, HD, scf, S, S, S, HD, false);
      softmax_causal<<<dim3(S), blk, 0, stream>>>(scf, pb, S, scale);
      const bf16* Bv = vT + ((long)(b * NKV + kv) * HD) * S;  // [HD, S]
      float* Catt = atf + (long)b * S * QD + (long)h * HD;
      gemm(pb, S, Bv, S, Catt, QD, S, HD, S, false);
    }
  }

  // 5) O projection -> d_out (overwrite)
  cvt(atf, atb, (long)T * QD);
  gemm(atb, QD, owb, QD, out, H, T, H, QD, false);

  // 6) SwiGLU MLP: h = silu(x gate^T) * (x up^T), then d_out += h down^T
  gemm_swiglu<<<dim3(IP / 128, T / 128), blk, 0, stream>>>(xb, H, gwb, uwb, H,
                                                           hb, IP, H);
  gemm(hb, IP, dwb, IP, out, H, T, H, IP, true);
}
